// HardQuadRadiusTripletLoss_40767829573849
// MI455X (gfx1250) — compile-verified, matrix operand order checked by reading, not run on hardware
//
#include <hip/hip_runtime.h>
#include <hip/hip_bf16.h>

typedef __attribute__((ext_vector_type(16))) __bf16 bf16x16;
typedef __attribute__((ext_vector_type(8)))  __bf16 bf16x8;
typedef __attribute__((ext_vector_type(8)))  float  f32x8;

#define GRID_SZ 8
#define B_   8
#define N_   2048
#define C_   256
#define H_   60
#define W_   60
#define HW_  3600
#define NTILES   (HW_ / 16)     // 225
#define NCHUNK   4              // HW split per keypoint tile
#define CHTILES  57             // ceil(225/4)
#define NEG_INIT (-3.0e38f)

// ---------------------------------------------------------------------------
// 1) desc2 (B,C,H,W) f32  ->  desc2^T (B,HW,C) bf16   (LDS tile transpose)
// ---------------------------------------------------------------------------
__global__ void k_transpose_bf16(const float* __restrict__ src, __bf16* __restrict__ dst) {
    __shared__ float tile[32][33];
    const int b  = blockIdx.z;
    const int c0 = blockIdx.y * 32;
    const int j0 = blockIdx.x * 32;
    const int tx = threadIdx.x, ty = threadIdx.y;   // 32 x 8
#pragma unroll
    for (int i = 0; i < 4; ++i) {
        int c = c0 + ty + i * 8;
        int j = j0 + tx;
        float v = 0.0f;
        if (j < HW_) v = src[(b * C_ + c) * HW_ + j];   // coalesced along j
        tile[ty + i * 8][tx] = v;
    }
    __syncthreads();
#pragma unroll
    for (int i = 0; i < 4; ++i) {
        int j = j0 + ty + i * 8;
        int c = c0 + tx;
        if (j < HW_) dst[(b * HW_ + j) * C_ + c] = (__bf16)tile[tx][ty + i * 8]; // coalesced along c
    }
}

// ---------------------------------------------------------------------------
// 2) kp1_desc f32 -> bf16 (same layout)
// ---------------------------------------------------------------------------
__global__ void k_convert_bf16(const float* __restrict__ src, __bf16* __restrict__ dst, int n) {
    int i = blockIdx.x * blockDim.x + threadIdx.x;
    if (i < n) dst[i] = (__bf16)src[i];
}

// ---------------------------------------------------------------------------
// 3) exact fp32 pos_sim: one wave32 per keypoint
// ---------------------------------------------------------------------------
__global__ void k_pos_sim(const float* __restrict__ wkp, const float* __restrict__ kdesc,
                          const float* __restrict__ desc2, float* __restrict__ pos) {
    const int wave = (blockIdx.x * blockDim.x + threadIdx.x) >> 5;
    const int lane = threadIdx.x & 31;
    const int b = wave / N_;
    float wy = wkp[wave * 2 + 0];
    float wx = wkp[wave * 2 + 1];
    int cy = (int)floorf(wy * (1.0f / GRID_SZ));
    int cx = (int)floorf(wx * (1.0f / GRID_SZ));
    cy = min(max(cy, 0), H_ - 1);
    cx = min(max(cx, 0), W_ - 1);
    const int idx = cy * W_ + cx;
    const float* kd = kdesc + (size_t)wave * C_;
    const float* d2 = desc2 + (size_t)b * C_ * HW_ + idx;
    float acc = 0.0f;
#pragma unroll
    for (int i = 0; i < C_ / 32; ++i) {
        int c = lane + i * 32;
        acc += kd[c] * d2[(size_t)c * HW_];
    }
#pragma unroll
    for (int off = 16; off > 0; off >>= 1)
        acc += __shfl_down(acc, off, 32);
    if (lane == 0) pos[wave] = acc;
}

// branchless insert into sorted-descending 4-list (compiler fuses to v_minmax_num)
#define TOP4_INSERT(t0, t1, t2, t3, v)                                         \
    do {                                                                       \
        float _u;                                                              \
        _u = fmaxf(t0, v); v = fminf(t0, v); t0 = _u;                          \
        _u = fmaxf(t1, v); v = fminf(t1, v); t1 = _u;                          \
        _u = fmaxf(t2, v); v = fminf(t2, v); t2 = _u;                          \
        t3 = fmaxf(t3, v);                                                     \
    } while (0)

// ---------------------------------------------------------------------------
// 4) fused GEMM (bf16 WMMA, f32 acc) + radius mask + running top-4.
//    Block = 8 waves sharing ONE (batch, hw-chunk): identical B-fragment
//    addresses in lock-step -> WGP$ hits for 7 of 8 waves. Waves differ
//    only in their 16-keypoint tile (A fragments, hoisted in VGPRs).
//    Grid: B * (N/16/8) * NCHUNK = 512 blocks, 4096 waves.
// ---------------------------------------------------------------------------
__global__ void __launch_bounds__(256, 1)
k_sim_topk(const __bf16* __restrict__ Abf, const __bf16* __restrict__ Btbf,
           const float* __restrict__ wkp, float* __restrict__ part) {
    __shared__ float smem[8][16][68];   // 68-float row stride: 16B aligned + conflict-free reads
    const int wvLocal = threadIdx.x >> 5;
    const int lane    = threadIdx.x & 31;
    const int s    = blockIdx.x & (NCHUNK - 1);     // hw chunk, shared by whole block
    const int bgrp = blockIdx.x >> 2;               // (b, nt-group)
    const int b    = bgrp >> 4;                     // / 16 groups per batch
    const int ntg  = bgrp & 15;
    const int nt   = ntg * 8 + wvLocal;             // this wave's keypoint tile
    const int n0   = nt * 16;
    const int col  = lane & 15;
    const int hi   = lane >> 4;

    // A fragments: 16 rows x K=256, hoisted across the whole hw loop.
    // 16-bit A layout: lanes 0-15 hold K pairs 0..7 & 16..23, lanes 16-31 hold 8..15 & 24..31.
    bf16x16 afrag[8];
    {
        const __bf16* abase = Abf + ((size_t)(b * N_ + n0 + col)) * C_ + hi * 8;
#pragma unroll
        for (int k = 0; k < 8; ++k) {
            const bf16x8* q = (const bf16x8*)(abase + k * 32);   // two b128 loads
            afrag[k] = __builtin_shufflevector(q[0], q[2],
                           0, 1, 2, 3, 4, 5, 6, 7, 8, 9, 10, 11, 12, 13, 14, 15);
        }
    }

    // w_kp1 for the 8 accumulator rows this lane holds (C/D layout: M = r + 8*hi)
    float wy[8], wx[8];
#pragma unroll
    for (int r = 0; r < 8; ++r) {
        int kp = b * N_ + n0 + hi * 8 + r;
        wy[r] = wkp[kp * 2 + 0];
        wx[r] = wkp[kp * 2 + 1];
    }

    float top[8][4];
#pragma unroll
    for (int r = 0; r < 8; ++r) {
        top[r][0] = NEG_INIT; top[r][1] = NEG_INIT;
        top[r][2] = NEG_INIT; top[r][3] = NEG_INIT;
    }

    const __bf16* bcol0 = Btbf + (size_t)b * HW_ * C_ + hi * 8;
    const int t0 = s * CHTILES;
    const int t1 = min(NTILES, t0 + CHTILES);

    for (int t = t0; t < t1; ++t) {
        const int j = t * 16 + col;                   // this lane's hw column
        const __bf16* bbase = bcol0 + (size_t)j * C_;
        const int h = j / W_;
        const int w = j - h * W_;
        const float yc = (h + 0.5f) * (float)GRID_SZ;
        const float xc = (w + 0.5f) * (float)GRID_SZ;

        f32x8 acc;
#pragma unroll
        for (int i = 0; i < 8; ++i) acc[i] = 0.0f;

#pragma unroll
        for (int k = 0; k < 8; ++k) {                 // K = 256 in chunks of 32
            const bf16x8* q = (const bf16x8*)(bbase + k * 32);
            bf16x16 bfrag = __builtin_shufflevector(q[0], q[2],
                                0, 1, 2, 3, 4, 5, 6, 7, 8, 9, 10, 11, 12, 13, 14, 15);
            acc = __builtin_amdgcn_wmma_f32_16x16x32_bf16(
                      false, afrag[k], false, bfrag, (short)0, acc, false, false);
        }
        // unconditional speculative prefetch of next tile's column (branch-free;
        // overrun stays inside the workspace allocation and is harmless)
        __builtin_prefetch(bbase + 16 * C_, 0, 1);    // global_prefetch_b8

        // mask (-1e4 inside radius) + top-4 insertion, all in registers
#pragma unroll
        for (int r = 0; r < 8; ++r) {
            float dy = wy[r] - yc;
            float dx = wx[r] - xc;
            float v  = (dy * dy + dx * dx <= 64.0f) ? -1.0e4f : acc[r];
            TOP4_INSERT(top[r][0], top[r][1], top[r][2], top[r][3], v);
        }
    }

    // cross-lane merge through LDS: 16 candidate columns x 4 per row
#pragma unroll
    for (int r = 0; r < 8; ++r) {
        int row = hi * 8 + r;
#pragma unroll
        for (int k = 0; k < 4; ++k)
            smem[wvLocal][row][col * 4 + k] = top[r][k];
    }
    __syncthreads();

    if (lane < 16) {
        const int row = lane;
        float m0 = NEG_INIT, m1 = NEG_INIT, m2 = NEG_INIT, m3 = NEG_INIT;
#pragma unroll 4
        for (int k = 0; k < 64; ++k) {
            float v = smem[wvLocal][row][k];
            TOP4_INSERT(m0, m1, m2, m3, v);
        }
        const int kp = b * N_ + n0 + row;
        float4* dst = (float4*)(part + ((size_t)kp * NCHUNK + s) * 4);
        *dst = make_float4(m0, m1, m2, m3);
    }
}

// ---------------------------------------------------------------------------
// 5) merge the NCHUNK partial top-4 lists, squared-hinge loss per keypoint
// ---------------------------------------------------------------------------
__global__ void k_merge_loss(const float* __restrict__ part, const float* __restrict__ pos,
                             float* __restrict__ lossPerKp) {
    int kp = blockIdx.x * blockDim.x + threadIdx.x;
    if (kp >= B_ * N_) return;
    float m0 = NEG_INIT, m1 = NEG_INIT, m2 = NEG_INIT, m3 = NEG_INIT;
    const float* p = part + (size_t)kp * (NCHUNK * 4);
#pragma unroll
    for (int i = 0; i < NCHUNK * 4; ++i) {
        float v = p[i];
        TOP4_INSERT(m0, m1, m2, m3, v);
    }
    const float pp = pos[kp];
    float l0 = fmaxf(m0 - pp + 1.0f, 0.0f);
    float l1 = fmaxf(m1 - pp + 1.0f, 0.0f);
    float l2 = fmaxf(m2 - pp + 1.0f, 0.0f);
    float l3 = fmaxf(m3 - pp + 1.0f, 0.0f);
    lossPerKp[kp] = l0 * l0 + l1 * l1 + l2 * l2 + l3 * l3;
}

// ---------------------------------------------------------------------------
// 6) deterministic fixed-order reduction to the scalar loss
// ---------------------------------------------------------------------------
__global__ void k_reduce(const float* __restrict__ lossPerKp, float* __restrict__ out) {
    __shared__ float sh[256];
    float acc = 0.0f;
    for (int i = threadIdx.x; i < B_ * N_; i += 256)
        acc += lossPerKp[i];
    sh[threadIdx.x] = acc;
    __syncthreads();
    for (int k = 128; k > 0; k >>= 1) {
        if ((int)threadIdx.x < k) sh[threadIdx.x] += sh[threadIdx.x + k];
        __syncthreads();
    }
    if (threadIdx.x == 0)
        out[0] = sh[0] * (1.0f / (float)(B_ * N_ * 4));   // mean over (B,N,num_neg), lambda=1
}

// ---------------------------------------------------------------------------
extern "C" void kernel_launch(void* const* d_in, const int* in_sizes, int n_in,
                              void* d_out, int out_size, void* d_ws, size_t ws_size,
                              hipStream_t stream) {
    (void)in_sizes; (void)n_in; (void)out_size; (void)ws_size;
    // inputs: 0=kp1 (unused), 1=w_kp1, 2=kp1_desc, 3=desc2, 4=homo12 (unused)
    const float* wkp   = (const float*)d_in[1];
    const float* kdesc = (const float*)d_in[2];
    const float* desc2 = (const float*)d_in[3];
    float* out = (float*)d_out;

    char* ws = (char*)d_ws;
    const size_t OFF_D2T  = 0;                         // 8*3600*256*2   = 14,745,600 B
    const size_t OFF_ABF  = OFF_D2T  + 14745600;       // 8*2048*256*2   =  8,388,608 B
    const size_t OFF_POS  = OFF_ABF  + 8388608;        // 16384*4        =     65,536 B
    const size_t OFF_LOSS = OFF_POS  + 65536;          // 16384*4        =     65,536 B
    const size_t OFF_PART = OFF_LOSS + 65536;          // 16384*16*4     =  1,048,576 B
    __bf16* d2t   = (__bf16*)(ws + OFF_D2T);
    __bf16* abf   = (__bf16*)(ws + OFF_ABF);
    float*  posb  = (float*)(ws + OFF_POS);
    float*  lossb = (float*)(ws + OFF_LOSS);
    float*  partb = (float*)(ws + OFF_PART);

    // 1) transpose+convert desc2 -> (B, HW, C) bf16
    dim3 ttb(32, 8, 1);
    dim3 ttg((HW_ + 31) / 32, C_ / 32, B_);
    k_transpose_bf16<<<ttg, ttb, 0, stream>>>(desc2, d2t);

    // 2) convert kp1_desc -> bf16
    const int nA = B_ * N_ * C_;
    k_convert_bf16<<<(nA + 255) / 256, 256, 0, stream>>>(kdesc, abf, nA);

    // 3) exact fp32 pos_sim (one wave per keypoint)
    k_pos_sim<<<(B_ * N_ * 32) / 256, 256, 0, stream>>>(wkp, kdesc, desc2, posb);

    // 4) fused WMMA correlation + mask + top-4 (4096 waves, 8 waves/block,
    //    block shares one (batch, hw-chunk) for WGP-cache reuse of B tiles)
    k_sim_topk<<<(B_ * (N_ / 16) * NCHUNK) / 8, 256, 0, stream>>>(abf, d2t, wkp, partb);

    // 5) merge partials + hinge^2 per keypoint
    k_merge_loss<<<(B_ * N_ + 255) / 256, 256, 0, stream>>>(partb, posb, lossb);

    // 6) deterministic scalar reduction
    k_reduce<<<1, 256, 0, stream>>>(lossb, out);
}
